// Stack_38448547234788
// MI455X (gfx1250) — compile-verified, matrix operand order checked by reading, not run on hardware
//
#include <hip/hip_runtime.h>
#include <hip/hip_bf16.h>

#define LN 64
#define MN 8
#define WN 2048
#define AN 128

// ---------- complex float helpers ----------
struct c32 { float x, y; };
__device__ __forceinline__ c32 cmk(float a, float b) { c32 r; r.x = a; r.y = b; return r; }
__device__ __forceinline__ c32 cadd(c32 a, c32 b) { return cmk(a.x + b.x, a.y + b.y); }
__device__ __forceinline__ c32 csub(c32 a, c32 b) { return cmk(a.x - b.x, a.y - b.y); }
__device__ __forceinline__ c32 cmul(c32 a, c32 b) { return cmk(a.x*b.x - a.y*b.y, a.x*b.y + a.y*b.x); }
__device__ __forceinline__ c32 cscale(c32 a, float s) { return cmk(a.x*s, a.y*s); }
__device__ __forceinline__ c32 cinv(c32 b) {
    float inv = __frcp_rn(b.x*b.x + b.y*b.y);
    return cmk(b.x*inv, -b.y*inv);
}
__device__ __forceinline__ c32 cdiv(c32 a, c32 b) { return cmul(a, cinv(b)); }
// principal branch complex sqrt (matches numpy/jax convention, y==+0 -> +i*sqrt(-x))
__device__ __forceinline__ c32 csqrtf(c32 z) {
    float r = sqrtf(z.x*z.x + z.y*z.y);
    float u = sqrtf(0.5f * fmaxf(r + z.x, 0.0f));
    float v = sqrtf(0.5f * fmaxf(r - z.x, 0.0f));
    v = (z.y < 0.0f) ? -v : v;
    return cmk(u, v);
}

// ---------- stage 1: one block per wavelength, one thread per angle ----------
__global__ __launch_bounds__(AN) void tmm_stage1(
    const float* __restrict__ thick,        // [LN]
    const float* __restrict__ n_real,       // [MN*WN]
    const float* __restrict__ n_imag,       // [MN*WN]
    const float* __restrict__ wavelengths,  // [WN]
    const float* __restrict__ angles,       // [AN]
    const int*   __restrict__ mat,          // [LN+2]
    float*       __restrict__ partials)     // [WN]
{
    __shared__ float2 s_n[LN + 2];  // complex refractive index per stack entry at this wavelength
    __shared__ float  s_c[LN];      // 2*pi*d_l / lambda per layer
    __shared__ float  s_red[AN];

    const int w = blockIdx.x;
    const int a = threadIdx.x;
    const float lam = wavelengths[w];

    if (a < LN + 2) {
        int m = mat[a];
        float nr = 1.0f + n_real[m * WN + w];
        float ni = 0.05f * n_imag[m * WN + w];
        if (a == 0) ni = 0.0f;  // lossless incident medium
        s_n[a] = make_float2(nr, ni);
    }
    if (a >= 64) {  // threads 64..127 stage the 64 per-layer phase coefficients
        int l = a - 64;
        float d_um = 0.05f + 0.45f * thick[l];  // microns
        s_c[l] = 2.0f * 3.14159265358979323846f * d_um / lam;
    }
    __syncthreads();

    const float sin0  = __sinf(angles[a]);
    const float alpha = s_n[0].x * sin0;  // n0*sin(theta0), real (n0 forced real)

    // nc_j = n_j * sqrt(1 - (alpha/n_j)^2), principal sqrt as in the reference
    auto nc_of = [&](int j) -> c32 {
        c32 nj = cmk(s_n[j].x, s_n[j].y);
        c32 s  = cmul(cmk(alpha, 0.0f), cinv(nj));
        c32 om = cmk(1.0f - (s.x*s.x - s.y*s.y), -2.0f * s.x * s.y);  // 1 - s*s
        return cmul(nj, csqrtf(om));
    };

    c32 nc_prev = nc_of(0);
    const float nc0_re = nc_prev.x;
    c32 nc_cur = nc_of(1);

    // M0 = [[1, r0],[r0, 1]] / t0
    {
        c32 sm = cadd(nc_prev, nc_cur);
        c32 r0 = cdiv(csub(nc_prev, nc_cur), sm);
        c32 it0 = cdiv(sm, cscale(nc_prev, 2.0f));  // 1/t0
        c32 M00v = it0;
        c32 M01v = cmul(r0, it0);
        c32 M00 = M00v, M01 = M01v, M10 = M01v, M11 = M00v;

        #pragma unroll 4
        for (int l = 0; l < LN; ++l) {
            c32 delta = cscale(nc_cur, s_c[l]);     // 2*pi*nc*d/lambda
            c32 nc_next = nc_of(l + 2);
            c32 sm2 = cadd(nc_cur, nc_next);
            c32 ri  = cdiv(csub(nc_cur, nc_next), sm2);
            c32 ti_inv = cdiv(sm2, cscale(nc_cur, 2.0f));  // 1/t_i

            float sc, cc;
            __sincosf(delta.x, &sc, &cc);
            float epos = __expf(delta.y);
            float eneg = __expf(-delta.y);
            c32 em = cmk(epos * cc, -epos * sc);  // exp(-i*delta)
            c32 ep = cmk(eneg * cc,  eneg * sc);  // exp(+i*delta)

            // Li = [[em, em*r],[ep*r, ep]] / t
            c32 L00 = cmul(em, ti_inv);
            c32 L01 = cmul(L00, ri);
            c32 L11 = cmul(ep, ti_inv);
            c32 L10 = cmul(L11, ri);

            c32 nM00 = cadd(cmul(M00, L00), cmul(M01, L10));
            c32 nM01 = cadd(cmul(M00, L01), cmul(M01, L11));
            c32 nM10 = cadd(cmul(M10, L00), cmul(M11, L10));
            c32 nM11 = cadd(cmul(M10, L01), cmul(M11, L11));
            M00 = nM00; M01 = nM01; M10 = nM10; M11 = nM11;
            nc_cur = nc_next;
        }

        c32 tt = cinv(M00);
        float T = (tt.x*tt.x + tt.y*tt.y) * (nc_cur.x / nc0_re);
        s_red[a] = 1.0f - T;
    }
    __syncthreads();
    #pragma unroll
    for (int s2 = AN / 2; s2 > 0; s2 >>= 1) {
        if (a < s2) s_red[a] += s_red[a + s2];
        __syncthreads();
    }
    if (a == 0) partials[w] = s_red[0];
}

// ---------- stage 2: single-wave WMMA reduction of 2048 partials ----------
// With an all-ones A matrix, sum over every element of D = 16 * sum over every
// element of B, independent of the VGPR striping permutation. So accumulating
// C += ones x B over all chunks and then summing all 256 accumulator values
// (8 VGPRs x 32 lanes) gives 16 * sum(partials).
typedef __attribute__((ext_vector_type(2)))  float    v2f;
typedef __attribute__((ext_vector_type(8)))  float    v8f;
typedef __attribute__((ext_vector_type(16))) _Float16 v16h;

__global__ __launch_bounds__(32) void tmm_reduce_wmma(
    const float* __restrict__ partials,  // [WN]
    float*       __restrict__ out)       // [1]
{
    const int lane = threadIdx.x;  // 0..31, one wave32, EXEC all ones
    v8f c = {};

#if __has_builtin(__builtin_amdgcn_wmma_f32_16x16x4_f32)
    v2f ones; ones[0] = 1.0f; ones[1] = 1.0f;
    // 16x16x4 f32: B is 4x16 = 64 floats = 2 VGPRs/lane. 2048/64 = 32 WMMAs.
    for (int it = 0; it < WN / 64; ++it) {
        const float* p = partials + it * 64;
        v2f b; b[0] = p[lane]; b[1] = p[lane + 32];
        c = __builtin_amdgcn_wmma_f32_16x16x4_f32(
                false, ones, false, b, (short)0, c, false, false);
    }
#else
    // fallback: codegen-confirmed f16 WMMA. B is 32x16 = 512 halves = 16/lane.
    v16h ones;
    #pragma unroll
    for (int k = 0; k < 16; ++k) ones[k] = (_Float16)1.0f;
    for (int it = 0; it < WN / 512; ++it) {
        const float* p = partials + it * 512 + lane * 16;
        v16h b;
        #pragma unroll
        for (int k = 0; k < 16; ++k) b[k] = (_Float16)p[k];
        c = __builtin_amdgcn_wmma_f32_16x16x32_f16(
                false, ones, false, b, (short)0, c, false, false);
    }
#endif

    float s = c[0] + c[1] + c[2] + c[3] + c[4] + c[5] + c[6] + c[7];
    #pragma unroll
    for (int off = 16; off > 0; off >>= 1) s += __shfl_down(s, off, 32);
    if (lane == 0) out[0] = (s / 16.0f) * (1.0f / (float)(WN * AN));
}

extern "C" void kernel_launch(void* const* d_in, const int* in_sizes, int n_in,
                              void* d_out, int out_size, void* d_ws, size_t ws_size,
                              hipStream_t stream) {
    const float* thick       = (const float*)d_in[0];  // [64]
    const float* n_real      = (const float*)d_in[1];  // [8*2048]
    const float* n_imag      = (const float*)d_in[2];  // [8*2048]
    const float* wavelengths = (const float*)d_in[3];  // [2048]
    const float* angles      = (const float*)d_in[4];  // [128]
    const int*   mat         = (const int*)d_in[5];    // [66]
    float* out = (float*)d_out;
    float* partials = (float*)d_ws;                    // needs WN*4 = 8 KB

    tmm_stage1<<<WN, AN, 0, stream>>>(thick, n_real, n_imag, wavelengths, angles, mat, partials);
    tmm_reduce_wmma<<<1, 32, 0, stream>>>(partials, out);
}
